// EnhancedDetector_59236188946840
// MI455X (gfx1250) — compile-verified, matrix operand order checked by reading, not run on hardware
//
#include <hip/hip_runtime.h>
#include <hip/hip_bf16.h>
#include <math.h>

// ---------------------------------------------------------------------------
// Problem constants (from the reference)
// ---------------------------------------------------------------------------
#define NN   50000
#define EE   800000
#define HH   512
#define GG   64
#define CC   2
#define BERT 768
#define XLD  (BERT + 1)   // x row stride = 769

typedef __attribute__((ext_vector_type(16))) __bf16 v16bf;
typedef __attribute__((ext_vector_type(16))) float  v16f;
typedef __attribute__((ext_vector_type(8)))  float  v8f;

__device__ __forceinline__ float gelu_exact(float x) {
    return 0.5f * x * (1.0f + erff(x * 0.70710678118654752f));
}

// ---------------------------------------------------------------------------
// Generic zero-fill
// ---------------------------------------------------------------------------
__global__ void k_zero(float* p, long n) {
    long i = (long)blockIdx.x * blockDim.x + threadIdx.x;
    if (i < n) p[i] = 0.0f;
}

// ---------------------------------------------------------------------------
// WMMA bf16 GEMM:  C[M x HH] = A[M x K] (f32, row stride LDA) @ W[K x HH]
//
// Block tile 128(M) x 64(N), K-step 32. 256 threads = 8 waves.
// Wave w computes rows [16w, 16w+16), all 64 columns (4 WMMA 16x16 tiles).
//
// LDS is FRAGMENT-MAJOR: one 32-byte slot per (strip/tile, lane) holding the
// 16 bf16 elements that lane feeds to v_wmma (ISA 16-bit A/B layout: half
// h = lane>>4 owns K runs [8h, 8h+7] and [16+8h, 16+8h+7]).  Staging swizzles
// on write (one 32B LDS store / slot); compute reads each fragment with one
// aligned 32B LDS load (2x ds_load_b128).
//
// LDA and the output width HH are compile-time constants so all strided
// addresses become base + constant immediate offsets (no per-pointer u64
// address chains).  Out-of-range A rows are CLAMPED to row 0 and loaded
// unconditionally; the garbage only reaches C rows >= M, which the guarded
// store never writes, so no value masking is needed.  VEC4=true uses float4
// (global_load_b128) loads for the two 8-float K-runs (requires LDA%4==0).
// ---------------------------------------------------------------------------
#define BM 128
#define BN 64
#define BK 32

template <int LDA, bool VEC4>
__global__ void k_wmma_gemm(const float* __restrict__ A,
                            const float* __restrict__ W,
                            const float* __restrict__ bias,   // per-column or nullptr
                            float* __restrict__ C,
                            int M, int K) {
    // 8 strips * 32 lanes * 16 bf16 = 8 KB ; 4 tiles * 32 lanes * 16 bf16 = 4 KB
    __shared__ alignas(32) __bf16 As[8 * 32 * 16];
    __shared__ alignas(32) __bf16 Bs[4 * 32 * 16];

    const int tid   = threadIdx.x;
    const int wave  = tid >> 5;
    const int lane  = tid & 31;
    const int half  = lane >> 4;   // 0: lanes 0-15, 1: lanes 16-31
    const int l15   = lane & 15;
    const int mBase = blockIdx.x * BM;
    const int nBase = blockIdx.y * BN;

    v8f acc[4] = {};

    // --- A staging role: thread `tid` owns A slot (strip=wave, lane=lane) ---
    const int  aRow  = mBase + wave * 16 + l15;
    const long aBase = (long)((aRow < M) ? aRow : 0) * LDA;   // clamped, legal
    const int  kA0   = 8 * half;        // first contiguous 8-K run
    const int  kA1   = 16 + 8 * half;   // second contiguous 8-K run

    // --- B staging role: threads 0..127 own B slot (tile=tid>>5, lane) ---
    const int  bCol  = nBase + ((tid >> 5) & 3) * 16 + l15;
    const int  kB0   = 8 * half;
    const int  kB1   = 16 + 8 * half;

    for (int k0 = 0; k0 < K; k0 += BK) {
        // ---- stage A slot: two contiguous 8-float runs, unconditional ----
        {
            v16f fa;
            if constexpr (VEC4) {
                const float4* p0 = reinterpret_cast<const float4*>(&A[aBase + k0 + kA0]);
                const float4* p1 = reinterpret_cast<const float4*>(&A[aBase + k0 + kA1]);
                float4 q0 = p0[0], q1 = p0[1];
                float4 q2 = p1[0], q3 = p1[1];
                fa[0]  = q0.x; fa[1]  = q0.y; fa[2]  = q0.z; fa[3]  = q0.w;
                fa[4]  = q1.x; fa[5]  = q1.y; fa[6]  = q1.z; fa[7]  = q1.w;
                fa[8]  = q2.x; fa[9]  = q2.y; fa[10] = q2.z; fa[11] = q2.w;
                fa[12] = q3.x; fa[13] = q3.y; fa[14] = q3.z; fa[15] = q3.w;
            } else {
                #pragma unroll
                for (int e = 0; e < 8; ++e) {
                    fa[e]     = A[aBase + (k0 + kA0 + e)];
                    fa[e + 8] = A[aBase + (k0 + kA1 + e)];
                }
            }
            if (k0 + BK < K)   // speculative prefetch of next K tile
                __builtin_prefetch(&A[aBase + (k0 + BK + kA0)], 0, 1);

            // single vector convert -> 8x v_cvt_pk_bf16_f32, no masking
            *reinterpret_cast<v16bf*>(&As[tid * 16]) =
                __builtin_convertvector(fa, v16bf);
        }
        // ---- stage B slot: 16 strided reads, constant immediate offsets ----
        if (tid < 128) {
            const float* wp = &W[(long)k0 * HH + bCol];
            v16f fb;
            #pragma unroll
            for (int e = 0; e < 8; ++e) {
                fb[e]     = wp[(kB0 + e) * HH];
                fb[e + 8] = wp[(kB1 + e) * HH];
            }
            *reinterpret_cast<v16bf*>(&Bs[tid * 16]) =
                __builtin_convertvector(fb, v16bf);
        }
        __syncthreads();

        // ---- fragment loads: one aligned 32B LDS read each ----
        v16bf afrag = *reinterpret_cast<const v16bf*>(&As[(wave * 32 + lane) * 16]);
        #pragma unroll
        for (int n = 0; n < 4; ++n) {
            v16bf bfrag = *reinterpret_cast<const v16bf*>(&Bs[(n * 32 + lane) * 16]);
            acc[n] = __builtin_amdgcn_wmma_f32_16x16x32_bf16(
                false, afrag, false, bfrag, (short)0, acc[n], false, false);
        }
        __syncthreads();
    }

    // store: C/D layout — VGPR r: lanes 0-15 -> M=r, lanes 16-31 -> M=8+r
    const bool hasBias = (bias != nullptr);
    #pragma unroll
    for (int n = 0; n < 4; ++n) {
        #pragma unroll
        for (int r = 0; r < 8; ++r) {
            int grow = mBase + wave * 16 + half * 8 + r;
            if (grow < M) {
                int gcol = nBase + n * 16 + l15;
                float v = acc[n][r];
                if (hasBias) v += bias[gcol];
                C[(long)grow * HH + gcol] = v;
            }
        }
    }
}

// ---------------------------------------------------------------------------
// Epilogue of the sem GEMM: h0 = gelu(LayerNorm(hw + slice_emb[idx]))
// One block (256 threads) per node row; each thread owns cols {t, t+256}.
// ---------------------------------------------------------------------------
__global__ void k_epi_ln_gelu(const float* __restrict__ hw,
                              const float* __restrict__ x,
                              const float* __restrict__ slice_emb,
                              const float* __restrict__ g,
                              const float* __restrict__ b,
                              float* __restrict__ out) {
    const int row = blockIdx.x;
    const int tid = threadIdx.x;
    __shared__ float rs[256];
    __shared__ float rq[256];

    const int idx = (int)x[(long)row * XLD + BERT];
    const long o  = (long)row * HH;

    float v0 = hw[o + tid]       + slice_emb[(long)idx * HH + tid];
    float v1 = hw[o + tid + 256] + slice_emb[(long)idx * HH + tid + 256];

    rs[tid] = v0 + v1;
    rq[tid] = v0 * v0 + v1 * v1;
    __syncthreads();
    for (int off = 128; off > 0; off >>= 1) {
        if (tid < off) { rs[tid] += rs[tid + off]; rq[tid] += rq[tid + off]; }
        __syncthreads();
    }
    float mean = rs[0] * (1.0f / HH);
    float var  = rq[0] * (1.0f / HH) - mean * mean;
    float rinv = rsqrtf(var + 1e-5f);

    out[o + tid]       = gelu_exact((v0 - mean) * rinv * g[tid]       + b[tid]);
    out[o + tid + 256] = gelu_exact((v1 - mean) * rinv * g[tid + 256] + b[tid + 256]);
}

// ---------------------------------------------------------------------------
// Degree: deg[dst] += 1 per edge (deg buffer pre-zeroed), then
// dinv = rsqrt(deg + 1) in place.
// ---------------------------------------------------------------------------
__global__ void k_deg_count(const int* __restrict__ ei, float* __restrict__ deg, int E) {
    int e = blockIdx.x * blockDim.x + threadIdx.x;
    if (e < E) atomicAdd(&deg[ei[E + e]], 1.0f);
}
__global__ void k_deg_finalize(float* __restrict__ deg, int n) {
    int i = blockIdx.x * blockDim.x + threadIdx.x;
    if (i < n) deg[i] = rsqrtf(deg[i] + 1.0f);
}

// ---------------------------------------------------------------------------
// agg = hw * dinv^2 + bias  (self-loop term + bias; scatter adds on top)
// One block per node row.
// ---------------------------------------------------------------------------
__global__ void k_agg_init(const float* __restrict__ hw,
                           const float* __restrict__ dinv,
                           const float* __restrict__ bias,
                           float* __restrict__ agg) {
    const long row = blockIdx.x;
    const int tid  = threadIdx.x;
    const float dd = dinv[row] * dinv[row];
    const long o   = row * HH;
    agg[o + tid]       = hw[o + tid]       * dd + bias[tid];
    agg[o + tid + 256] = hw[o + tid + 256] * dd + bias[tid + 256];
}

// ---------------------------------------------------------------------------
// Edge scatter: agg[dst] += hw[src] * dinv[src]*dinv[dst].  One block/edge.
// ---------------------------------------------------------------------------
__global__ void k_edge_scatter(const int* __restrict__ ei,
                               const float* __restrict__ dinv,
                               const float* __restrict__ hw,
                               float* __restrict__ agg, int E) {
    const int e   = blockIdx.x;
    const int tid = threadIdx.x;
    const int src = ei[e];
    const int dst = ei[E + e];
    const float coef = dinv[src] * dinv[dst];
    const long so = (long)src * HH;
    const long dofs = (long)dst * HH;
    atomicAdd(&agg[dofs + tid],       hw[so + tid]       * coef);
    atomicAdd(&agg[dofs + tid + 256], hw[so + tid + 256] * coef);
}

__global__ void k_gelu_inplace(float* __restrict__ p, long n) {
    long i = (long)blockIdx.x * blockDim.x + threadIdx.x;
    if (i < n) p[i] = gelu_exact(p[i]);
}

// ---------------------------------------------------------------------------
// Mean-pool accumulation: gsum[batch[i]] += h1[i]+h2[i]; gcnt[batch[i]] += 1
// ---------------------------------------------------------------------------
__global__ void k_pool(const float* __restrict__ h1, const float* __restrict__ h2,
                       const int* __restrict__ batch,
                       float* __restrict__ gsum, float* __restrict__ gcnt) {
    const int row = blockIdx.x;
    const int tid = threadIdx.x;
    const int g   = batch[row];
    const long o  = (long)row * HH;
    atomicAdd(&gsum[(long)g * HH + tid],       h1[o + tid]       + h2[o + tid]);
    atomicAdd(&gsum[(long)g * HH + tid + 256], h1[o + tid + 256] + h2[o + tid + 256]);
    if (tid == 0) atomicAdd(&gcnt[g], 1.0f);
}

// ---------------------------------------------------------------------------
// Classifier: hg = gsum/max(cnt,1); z = gelu(hg@W1+b1); out = z@W2+b2
// One block per graph (64 blocks, 256 threads).
// ---------------------------------------------------------------------------
__global__ void k_classifier(const float* __restrict__ gsum,
                             const float* __restrict__ gcnt,
                             const float* __restrict__ W1, const float* __restrict__ b1,
                             const float* __restrict__ W2, const float* __restrict__ b2,
                             float* __restrict__ out) {
    const int g   = blockIdx.x;
    const int tid = threadIdx.x;
    __shared__ float hg[HH];
    __shared__ float z[HH];
    __shared__ float red[256];

    const float cnt = fmaxf(gcnt[g], 1.0f);
    hg[tid]       = gsum[(long)g * HH + tid]       / cnt;
    hg[tid + 256] = gsum[(long)g * HH + tid + 256] / cnt;
    __syncthreads();

    for (int j = tid; j < HH; j += 256) {
        float acc = b1[j];
        for (int k = 0; k < HH; ++k) acc += hg[k] * W1[(long)k * HH + j];
        z[j] = gelu_exact(acc);
    }
    __syncthreads();

    for (int c = 0; c < CC; ++c) {
        float p = 0.0f;
        for (int j = tid; j < HH; j += 256) p += z[j] * W2[(long)j * CC + c];
        red[tid] = p;
        __syncthreads();
        for (int off = 128; off > 0; off >>= 1) {
            if (tid < off) red[tid] += red[tid + off];
            __syncthreads();
        }
        if (tid == 0) out[g * CC + c] = red[0] + b2[c];
        __syncthreads();
    }
}

// ---------------------------------------------------------------------------
// Host-side orchestration
// ---------------------------------------------------------------------------
extern "C" void kernel_launch(void* const* d_in, const int* in_sizes, int n_in,
                              void* d_out, int out_size, void* d_ws, size_t ws_size,
                              hipStream_t stream) {
    (void)in_sizes; (void)n_in; (void)out_size; (void)ws_size;

    const float* x        = (const float*)d_in[0];
    const int*   ei       = (const int*)  d_in[1];
    const int*   batch    = (const int*)  d_in[2];
    const float* sem_W    = (const float*)d_in[3];
    const float* sem_b    = (const float*)d_in[4];
    const float* slice_e  = (const float*)d_in[5];
    const float* ln_g     = (const float*)d_in[6];
    const float* ln_b     = (const float*)d_in[7];
    const float* conv1_W  = (const float*)d_in[8];
    const float* conv1_b  = (const float*)d_in[9];
    const float* conv2_W  = (const float*)d_in[10];
    const float* conv2_b  = (const float*)d_in[11];
    const float* cls_W1   = (const float*)d_in[12];
    const float* cls_b1   = (const float*)d_in[13];
    const float* cls_W2   = (const float*)d_in[14];
    const float* cls_b2   = (const float*)d_in[15];
    float* out = (float*)d_out;

    const long NH = (long)NN * HH;
    float* f0   = (float*)d_ws;      // h0 / agg2 / h2
    float* f1   = f0 + NH;           // hw scratch
    float* f2   = f1 + NH;           // agg1 / h1
    float* dinv = f2 + NH;           // N
    float* gsum = dinv + NN;         // G*H
    float* gcnt = gsum + (long)GG * HH; // G

    const dim3 blk(256);
    const dim3 gemmGrid((NN + BM - 1) / BM, HH / BN);

    // zero: dinv (deg accum), gsum, gcnt (contiguous in ws)
    {
        long nz = NN + (long)GG * HH + GG;
        k_zero<<<dim3((nz + 255) / 256), blk, 0, stream>>>(dinv, nz);
    }

    // 1) hw = sem @ sem_W + sem_b     (A = x, row stride 769 -> scalar loads)
    k_wmma_gemm<XLD, false><<<gemmGrid, blk, 0, stream>>>(x, sem_W, sem_b, f1, NN, BERT);
    // 2) h0 = gelu(LN(hw + slice_emb[idx]))
    k_epi_ln_gelu<<<dim3(NN), blk, 0, stream>>>(f1, x, slice_e, ln_g, ln_b, f0);

    // 3) degree -> dinv
    k_deg_count<<<dim3((EE + 255) / 256), blk, 0, stream>>>(ei, dinv, EE);
    k_deg_finalize<<<dim3((NN + 255) / 256), blk, 0, stream>>>(dinv, NN);

    // 4) conv1: hw = h0 @ conv1_W ; agg = hw*dinv^2 + b ; scatter ; gelu
    k_wmma_gemm<HH, true><<<gemmGrid, blk, 0, stream>>>(f0, conv1_W, nullptr, f1, NN, HH);
    k_agg_init<<<dim3(NN), blk, 0, stream>>>(f1, dinv, conv1_b, f2);
    k_edge_scatter<<<dim3(EE), blk, 0, stream>>>(ei, dinv, f1, f2, EE);
    k_gelu_inplace<<<dim3((NH + 255) / 256), blk, 0, stream>>>(f2, NH);   // f2 = h1

    // 5) conv2: hw2 = h1 @ conv2_W ; agg2 ; scatter ; gelu
    k_wmma_gemm<HH, true><<<gemmGrid, blk, 0, stream>>>(f2, conv2_W, nullptr, f1, NN, HH);
    k_agg_init<<<dim3(NN), blk, 0, stream>>>(f1, dinv, conv2_b, f0);
    k_edge_scatter<<<dim3(EE), blk, 0, stream>>>(ei, dinv, f1, f0, EE);
    k_gelu_inplace<<<dim3((NH + 255) / 256), blk, 0, stream>>>(f0, NH);   // f0 = h2

    // 6) mean pool (h1 + h2) per graph
    k_pool<<<dim3(NN), blk, 0, stream>>>(f2, f0, batch, gsum, gcnt);

    // 7) classifier
    k_classifier<<<dim3(GG), blk, 0, stream>>>(gsum, gcnt, cls_W1, cls_b1,
                                               cls_W2, cls_b2, out);
}
